// GraphFusion_77704548319488
// MI455X (gfx1250) — compile-verified, matrix operand order checked by reading, not run on hardware
//
#include <hip/hip_runtime.h>
#include <hip/hip_bf16.h>

typedef __attribute__((ext_vector_type(16))) _Float16 v16h;
typedef __attribute__((ext_vector_type(8)))  _Float16 v8h;
typedef __attribute__((ext_vector_type(8)))  float    v8f;

#define D      128
#define HID    64
#define TILE_M 16

#define DEV __device__ __forceinline__

// d_ws layout (offsets in _Float16 elements): transposed f16 weights [N][K]
#define OFF_GF_W1   0        // [64][256]
#define OFF_GF_W2   16384    // [128][64]
#define OFF_GF2_W1  24576    // [64][256]
#define OFF_GF2_W2  40960    // [128][64]
#define OFF_LL_W1   49152    // [64][384]
#define OFF_LL_W2   73728    // [64][64]
#define OFF_LL_W3   77824    // [128][64]

constexpr int ACT_LRELU = 0;
constexpr int ACT_TANH  = 1;

struct __align__(16) Smem {
  _Float16 smX[3][TILE_M][D];     // softmaxed a, v, l           (12288 B)
  _Float16 G[3][TILE_M][D];       // gav, gal, gvl               (12288 B)
  _Float16 G2[3][TILE_M][D];      // softmax(g*) -> a_v,a_l,v_l  (12288 B)
  _Float16 fusion[TILE_M][3 * D]; // uni | bim | tri             (12288 B)
  _Float16 hbuf[TILE_M][HID];     // hidden staging              ( 2048 B)
  _Float16 hbuf2[TILE_M][HID];    //                             ( 2048 B)
  float    tri[TILE_M][D];        // trimodal f32 accumulator    ( 8192 B)
  float    attw[D];               //                             (  512 B)
  float    rowscal[TILE_M][8];    // sa sv sl sav sal svl        (  512 B)
  float    norm[TILE_M][4];
  float    n2[TILE_M][8];
};

DEV v8f zero8() {
  v8f z;
#pragma unroll
  for (int j = 0; j < 8; ++j) z[j] = 0.f;
  return z;
}

DEV v16h load_frag(const _Float16* p) {
  v8h lo = *reinterpret_cast<const v8h*>(p);
  v8h hi = *reinterpret_cast<const v8h*>(p + 16);
  return __builtin_shufflevector(lo, hi, 0, 1, 2, 3, 4, 5, 6, 7,
                                 8, 9, 10, 11, 12, 13, 14, 15);
}

// C[NT] (16x16 f32 tiles) += A(16 x 32*KT f16, row-major ldA) * Wt[n][k] slice.
// A-fragment (16-bit A 16x32): lane<16 -> K {0..7,16..23}, lane>=16 -> +8.
// B mirrors A with N in place of M; Wt stored [N][K] so both fragments are
// two contiguous 16B loads per lane.  Fully unrolled: immediate offsets.
template <int NT, int KT>
DEV void mma_block(v8f* C, const _Float16* __restrict__ A, int ldA,
                   const _Float16* __restrict__ Wt, int KW, int kBase, int lane) {
  const int lr = lane & 15;
  const int lh = lane >> 4;
  const _Float16* ap = A + lr * ldA + lh * 8;
  const _Float16* bp = Wt + lr * KW + kBase + lh * 8;
#pragma unroll
  for (int kt = 0; kt < KT; ++kt) {
    const int kOff = kt * 32;
    v16h a = load_frag(ap + kOff);
#pragma unroll
    for (int nt = 0; nt < NT; ++nt) {
      v16h b = load_frag(bp + nt * 16 * KW + kOff);
      C[nt] = __builtin_amdgcn_wmma_f32_16x16x32_f16(
          false, a, false, b, (short)0, C[nt], false, false);
    }
  }
}

// C/D layout: N = lane&15 (per 16-col tile), M = j + 8*(lane>>4)
template <int NT, int ACT>
DEV void epi_store_h(const v8f* C, const float* __restrict__ bias,
                     _Float16* dst, int ld, int lane) {
  const int lr = lane & 15, lh = lane >> 4;
#pragma unroll
  for (int nt = 0; nt < NT; ++nt) {
    const int col = nt * 16 + lr;
    const float bb = bias[col];
#pragma unroll
    for (int j = 0; j < 8; ++j) {
      float v = C[nt][j] + bb;
      v = (ACT == ACT_LRELU) ? (v > 0.f ? v : 0.2f * v) : tanhf(v);
      dst[(8 * lh + j) * ld + col] = (_Float16)v;
    }
  }
}

DEV float rowdot_h(const _Float16* X, const _Float16* Y, int r, int c0) {
  const _Float16* xp = X + r * D + c0;
  const _Float16* yp = Y + r * D + c0;
  float d = 0.f;
#pragma unroll
  for (int i = 0; i < 64; ++i) d += (float)xp[i] * (float)yp[i];
  d += __shfl_xor(d, 1, 32);
  return d;
}

// One gf/gf2 evaluation for the wave's 16-row tile:
//   h = lrelu(concat(X,Y) @ W1 + b1); g = tanh(h @ W2 + b2)
// triIdx < 0: store g (f16) to outG.  triIdx >= 0: tri += tanh(n2[:,triIdx]*g).
DEV void gf_call(Smem& s, const _Float16* Xh, const _Float16* Yh,
                 const _Float16* __restrict__ w1t, const float* __restrict__ b1,
                 const _Float16* __restrict__ w2t, const float* __restrict__ b2,
                 _Float16* outG, int triIdx, int lane) {
  v8f C1[4];
#pragma unroll
  for (int nt = 0; nt < 4; ++nt) C1[nt] = zero8();
  mma_block<4, 4>(C1, Xh, D, w1t, 2 * D, 0, lane);
  mma_block<4, 4>(C1, Yh, D, w1t, 2 * D, D, lane);
  epi_store_h<4, ACT_LRELU>(C1, b1, &s.hbuf[0][0], HID, lane);
  __syncthreads();

  v8f C2[8];
#pragma unroll
  for (int nt = 0; nt < 8; ++nt) C2[nt] = zero8();
  mma_block<8, 2>(C2, &s.hbuf[0][0], HID, w2t, HID, 0, lane);

  const int lr = lane & 15, lh = lane >> 4;
  if (triIdx < 0) {
    epi_store_h<8, ACT_TANH>(C2, b2, outG, D, lane);
  } else {
#pragma unroll
    for (int nt = 0; nt < 8; ++nt) {
      const int col = nt * 16 + lr;
      const float bb = b2[col];
#pragma unroll
      for (int j = 0; j < 8; ++j) {
        const int row = 8 * lh + j;
        const float g = tanhf(C2[nt][j] + bb);
        s.tri[row][col] += tanhf(s.n2[row][triIdx] * g);
      }
    }
  }
  __syncthreads();
}

__global__ void prep_transpose(const float* __restrict__ w, _Float16* __restrict__ o,
                               int K, int N) {
  int i = blockIdx.x * blockDim.x + threadIdx.x;
  if (i < K * N) {
    int n = i / K, k = i - n * K;
    o[i] = (_Float16)w[(size_t)k * N + n];  // o[n*K + k] = w[k][n]
  }
}

__global__ __launch_bounds__(32) __attribute__((flatten)) void fusion_main(
    const float* __restrict__ Lp, const float* __restrict__ Ap,
    const float* __restrict__ Vp, const float* __restrict__ att_w,
    const float* __restrict__ att_b, const float* __restrict__ gf_b1,
    const float* __restrict__ gf_b2, const float* __restrict__ gf2_b1,
    const float* __restrict__ gf2_b2, const float* __restrict__ ll_b1,
    const float* __restrict__ ll_b2, const float* __restrict__ ll_b3,
    const _Float16* __restrict__ wh, float* __restrict__ out) {
  __shared__ Smem s;
  const int lane = threadIdx.x;
  const int base = blockIdx.x * TILE_M;
  const int r = lane >> 1;          // 0..15: row for elementwise phases
  const int c0 = (lane & 1) * 64;   // half-row of 64 cols

  for (int i = lane; i < D; i += 32) s.attw[i] = att_w[i];
  __syncthreads();
  const float attb = att_b[0];

  // -------- Phase 1: attention scalars, unimodal, row softmaxes (a, v, l) ----
  const float* Xsrc[3] = {Ap, Vp, Lp};
  float uni[64];
#pragma unroll
  for (int i = 0; i < 64; ++i) uni[i] = 0.f;

#pragma unroll
  for (int m = 0; m < 3; ++m) {
    const float* src = Xsrc[m] + (size_t)(base + r) * D + c0;
    float x[64];
#pragma unroll
    for (int i = 0; i < 64; ++i) x[i] = src[i];
    float dsc = 0.f;
#pragma unroll
    for (int i = 0; i < 64; ++i) dsc += x[i] * s.attw[c0 + i];
    dsc += __shfl_xor(dsc, 1, 32);
    const float sm = tanhf(dsc + attb);
    if ((lane & 1) == 0) s.rowscal[r][m] = sm;
#pragma unroll
    for (int i = 0; i < 64; ++i) uni[i] += sm * x[i];
    float mx = -3.4e38f;
#pragma unroll
    for (int i = 0; i < 64; ++i) mx = fmaxf(mx, x[i]);
    mx = fmaxf(mx, __shfl_xor(mx, 1, 32));
    float ssum = 0.f;
#pragma unroll
    for (int i = 0; i < 64; ++i) { x[i] = expf(x[i] - mx); ssum += x[i]; }
    ssum += __shfl_xor(ssum, 1, 32);
    const float inv = 1.f / ssum;
#pragma unroll
    for (int i = 0; i < 64; ++i) s.smX[m][r][c0 + i] = (_Float16)(x[i] * inv);
  }
#pragma unroll
  for (int i = 0; i < 64; ++i) s.fusion[r][c0 + i] = (_Float16)(uni[i] * (1.f / 3.f));
  __syncthreads();

  // -------- Phase 2: sav/sal/svl + normalize (softmax over 3) ---------------
  {
    const float dav = rowdot_h(&s.smX[0][0][0], &s.smX[1][0][0], r, c0);
    const float dal = rowdot_h(&s.smX[0][0][0], &s.smX[2][0][0], r, c0);
    const float dvl = rowdot_h(&s.smX[1][0][0], &s.smX[2][0][0], r, c0);
    const float sa = s.rowscal[r][0], sv = s.rowscal[r][1], sl = s.rowscal[r][2];
    const float sav = (sa + sv) / (dav + 0.5f);
    const float sal = (sa + sl) / (dal + 0.5f);
    const float svl = (sl + sv) / (dvl + 0.5f);
    if ((lane & 1) == 0) {
      s.rowscal[r][3] = sav; s.rowscal[r][4] = sal; s.rowscal[r][5] = svl;
    }
  }
  __syncthreads();
  if (lane < TILE_M) {
    const float sav = s.rowscal[lane][3], sal = s.rowscal[lane][4],
                svl = s.rowscal[lane][5];
    const float mx = fmaxf(sav, fmaxf(sal, svl));
    const float e0 = expf(sav - mx), e1 = expf(sal - mx), e2 = expf(svl - mx);
    const float inv = 1.f / (e0 + e1 + e2);
    s.norm[lane][0] = e0 * inv; s.norm[lane][1] = e1 * inv; s.norm[lane][2] = e2 * inv;
  }
  __syncthreads();

  const _Float16* w_gf1  = wh + OFF_GF_W1;
  const _Float16* w_gf2  = wh + OFF_GF_W2;
  const _Float16* w_gf21 = wh + OFF_GF2_W1;
  const _Float16* w_gf22 = wh + OFF_GF2_W2;

  // -------- Phase 3: gav, gal, gvl via WMMA ---------------------------------
  gf_call(s, &s.smX[0][0][0], &s.smX[1][0][0], w_gf1, gf_b1, w_gf2, gf_b2,
          &s.G[0][0][0], -1, lane);
  gf_call(s, &s.smX[0][0][0], &s.smX[2][0][0], w_gf1, gf_b1, w_gf2, gf_b2,
          &s.G[1][0][0], -1, lane);
  gf_call(s, &s.smX[1][0][0], &s.smX[2][0][0], w_gf1, gf_b1, w_gf2, gf_b2,
          &s.G[2][0][0], -1, lane);

  // -------- Phase 4: row softmax of each g ----------------------------------
#pragma unroll
  for (int gi = 0; gi < 3; ++gi) {
    float x[64];
#pragma unroll
    for (int i = 0; i < 64; ++i) x[i] = (float)s.G[gi][r][c0 + i];
    float mx = -3.4e38f;
#pragma unroll
    for (int i = 0; i < 64; ++i) mx = fmaxf(mx, x[i]);
    mx = fmaxf(mx, __shfl_xor(mx, 1, 32));
    float ssum = 0.f;
#pragma unroll
    for (int i = 0; i < 64; ++i) { x[i] = expf(x[i] - mx); ssum += x[i]; }
    ssum += __shfl_xor(ssum, 1, 32);
    const float inv = 1.f / ssum;
#pragma unroll
    for (int i = 0; i < 64; ++i) s.G2[gi][r][c0 + i] = (_Float16)(x[i] * inv);
  }
  __syncthreads();

  // -------- Phase 5: six pairwise dots -> n2 (softmax over 6) ---------------
  {
    const float d1 = rowdot_h(&s.G2[0][0][0], &s.G2[2][0][0], r, c0);  // av2.vl2
    const float d2 = rowdot_h(&s.G2[0][0][0], &s.G2[1][0][0], r, c0);  // av2.al2
    const float d3 = rowdot_h(&s.G2[1][0][0], &s.G2[2][0][0], r, c0);  // al2.vl2
    const float d4 = rowdot_h(&s.G2[0][0][0], &s.smX[2][0][0], r, c0); // av2.l
    const float d5 = rowdot_h(&s.G2[1][0][0], &s.smX[1][0][0], r, c0); // al2.v
    const float d6 = rowdot_h(&s.G2[2][0][0], &s.smX[0][0][0], r, c0); // vl2.a
    const float sa = s.rowscal[r][0], sv = s.rowscal[r][1], sl = s.rowscal[r][2];
    const float sav = s.rowscal[r][3], sal = s.rowscal[r][4], svl = s.rowscal[r][5];
    float t[6];
    t[0] = (sav + svl) / (d1 + 0.5f);
    t[1] = (sav + sal) / (d2 + 0.5f);
    t[2] = (sal + svl) / (d3 + 0.5f);
    t[3] = (sav + sl) / (d4 + 0.5f);
    t[4] = (sal + sv) / (d5 + 0.5f);
    t[5] = (sa + svl) / (d6 + 0.5f);
    float mx = t[0];
#pragma unroll
    for (int k = 1; k < 6; ++k) mx = fmaxf(mx, t[k]);
    float ssum = 0.f;
#pragma unroll
    for (int k = 0; k < 6; ++k) { t[k] = expf(t[k] - mx); ssum += t[k]; }
    const float inv = 1.f / ssum;
    if ((lane & 1) == 0) {
#pragma unroll
      for (int k = 0; k < 6; ++k) s.n2[r][k] = t[k] * inv;
    }
  }
  __syncthreads();

  // -------- Phase 6: a_v/a_l/v_l, bimodal, zero trimodal --------------------
  {
    const float n0 = s.norm[r][0], n1 = s.norm[r][1], n2c = s.norm[r][2];
#pragma unroll
    for (int i = 0; i < 64; ++i) {
      const float av = tanhf(n0 * (float)s.G[0][r][c0 + i]);
      const float al = tanhf(n1 * (float)s.G[1][r][c0 + i]);
      const float vl = tanhf(n2c * (float)s.G[2][r][c0 + i]);
      s.G2[0][r][c0 + i] = (_Float16)av;
      s.G2[1][r][c0 + i] = (_Float16)al;
      s.G2[2][r][c0 + i] = (_Float16)vl;
      s.fusion[r][D + c0 + i] = (_Float16)(av + al + vl);
      s.tri[r][c0 + i] = 0.f;
    }
  }
  __syncthreads();

  // -------- Phase 7: trimodal = sum of 6 gf2 terms (WMMA) -------------------
  gf_call(s, &s.G2[0][0][0], &s.G2[2][0][0], w_gf21, gf2_b1, w_gf22, gf2_b2, nullptr, 0, lane);
  gf_call(s, &s.G2[0][0][0], &s.G2[1][0][0], w_gf21, gf2_b1, w_gf22, gf2_b2, nullptr, 1, lane);
  gf_call(s, &s.G2[2][0][0], &s.G2[1][0][0], w_gf21, gf2_b1, w_gf22, gf2_b2, nullptr, 2, lane);
  gf_call(s, &s.G2[0][0][0], &s.smX[2][0][0], w_gf21, gf2_b1, w_gf22, gf2_b2, nullptr, 3, lane);
  gf_call(s, &s.G2[1][0][0], &s.smX[1][0][0], w_gf21, gf2_b1, w_gf22, gf2_b2, nullptr, 4, lane);
  gf_call(s, &s.G2[2][0][0], &s.smX[0][0][0], w_gf21, gf2_b1, w_gf22, gf2_b2, nullptr, 5, lane);
#pragma unroll
  for (int i = 0; i < 64; ++i)
    s.fusion[r][2 * D + c0 + i] = (_Float16)s.tri[r][c0 + i];
  __syncthreads();

  // -------- Phase 8: head 384 -> 64 -> 64 -> 128 (WMMA) ---------------------
  {
    v8f C1[4];
#pragma unroll
    for (int nt = 0; nt < 4; ++nt) C1[nt] = zero8();
    mma_block<4, 12>(C1, &s.fusion[0][0], 3 * D, wh + OFF_LL_W1, 3 * D, 0, lane);
    epi_store_h<4, ACT_TANH>(C1, ll_b1, &s.hbuf[0][0], HID, lane);
    __syncthreads();

    v8f C2a[4];
#pragma unroll
    for (int nt = 0; nt < 4; ++nt) C2a[nt] = zero8();
    mma_block<4, 2>(C2a, &s.hbuf[0][0], HID, wh + OFF_LL_W2, HID, 0, lane);
    epi_store_h<4, ACT_TANH>(C2a, ll_b2, &s.hbuf2[0][0], HID, lane);
    __syncthreads();

    v8f C3[8];
#pragma unroll
    for (int nt = 0; nt < 8; ++nt) C3[nt] = zero8();
    mma_block<8, 2>(C3, &s.hbuf2[0][0], HID, wh + OFF_LL_W3, HID, 0, lane);
    const int lr = lane & 15, lh = lane >> 4;
#pragma unroll
    for (int nt = 0; nt < 8; ++nt) {
      const int col = nt * 16 + lr;
      const float bb = ll_b3[col];
#pragma unroll
      for (int j = 0; j < 8; ++j) {
        const int row = 8 * lh + j;
        out[(size_t)(base + row) * D + col] = tanhf(C3[nt][j] + bb);
      }
    }
  }
}

extern "C" void kernel_launch(void* const* d_in, const int* in_sizes, int n_in,
                              void* d_out, int out_size, void* d_ws, size_t ws_size,
                              hipStream_t stream) {
  const float* Lp     = (const float*)d_in[0];
  const float* Ap     = (const float*)d_in[1];
  const float* Vp     = (const float*)d_in[2];
  const float* att_w  = (const float*)d_in[3];
  const float* att_b  = (const float*)d_in[4];
  const float* gf_w1  = (const float*)d_in[5];
  const float* gf_b1  = (const float*)d_in[6];
  const float* gf_w2  = (const float*)d_in[7];
  const float* gf_b2  = (const float*)d_in[8];
  const float* gf2_w1 = (const float*)d_in[9];
  const float* gf2_b1 = (const float*)d_in[10];
  const float* gf2_w2 = (const float*)d_in[11];
  const float* gf2_b2 = (const float*)d_in[12];
  const float* ll_w1  = (const float*)d_in[13];
  const float* ll_b1  = (const float*)d_in[14];
  const float* ll_w2  = (const float*)d_in[15];
  const float* ll_b2  = (const float*)d_in[16];
  const float* ll_w3  = (const float*)d_in[17];
  const float* ll_b3  = (const float*)d_in[18];

  _Float16* wh = (_Float16*)d_ws;  // ~172 KB of transposed f16 weights

  auto prep = [&](const float* w, int off, int K, int N) {
    const int n = K * N;
    prep_transpose<<<(n + 255) / 256, 256, 0, stream>>>(w, wh + off, K, N);
  };
  prep(gf_w1,  OFF_GF_W1,  256, 64);
  prep(gf_w2,  OFF_GF_W2,  64, 128);
  prep(gf2_w1, OFF_GF2_W1, 256, 64);
  prep(gf2_w2, OFF_GF2_W2, 64, 128);
  prep(ll_w1,  OFF_LL_W1,  384, 64);
  prep(ll_w2,  OFF_LL_W2,  64, 64);
  prep(ll_w3,  OFF_LL_W3,  64, 128);

  const int B = in_sizes[0] / D;  // 65536
  fusion_main<<<B / TILE_M, 32, 0, stream>>>(
      Lp, Ap, Vp, att_w, att_b, gf_b1, gf_b2, gf2_b1, gf2_b2,
      ll_b1, ll_b2, ll_b3, wh, (float*)d_out);
}
